// graphCNN_49632642072642
// MI455X (gfx1250) — matrix operand order repaired and checked by reading, not compiled, not run on hardware
//
#include <hip/hip_runtime.h>
#include <hip/hip_bf16.h>

typedef __attribute__((ext_vector_type(16))) _Float16 v16h;
typedef __attribute__((ext_vector_type(8)))  _Float16 v8h;
typedef __attribute__((ext_vector_type(8)))  float    v8f;

__device__ __forceinline__ float sigmf(float x) { return 1.0f / (1.0f + __expf(-x)); }

// ---------------------------------------------------------------------------
// Conv2d(12->12, k=4, s=2, p=1) + bias. in (512,12,24,72) -> y (512,12,12,36)
// ---------------------------------------------------------------------------
__global__ void conv_bias_kernel(const float* __restrict__ x,
                                 const float* __restrict__ W,   // (12,12,4,4)
                                 const float* __restrict__ bias,
                                 float* __restrict__ y) {
  __shared__ float sW[12 * 12 * 16];
  for (int i = threadIdx.x; i < 2304; i += blockDim.x) sW[i] = W[i];
  __syncthreads();
  int idx = blockIdx.x * blockDim.x + threadIdx.x;
  const int total = 512 * 12 * 12 * 36;
  if (idx >= total) return;
  int ow = idx % 36; int t = idx / 36;
  int oh = t % 12;   t /= 12;
  int co = t % 12;   int b = t / 12;
  float acc = bias[co];
  const float* xb = x + (size_t)b * (12 * 24 * 72);
  const float* wc = sW + co * (12 * 16);
  for (int ci = 0; ci < 12; ++ci) {
    const float* xc = xb + ci * (24 * 72);
    const float* wk = wc + ci * 16;
#pragma unroll
    for (int kh = 0; kh < 4; ++kh) {
      int ih = oh * 2 - 1 + kh;
      if (ih < 0 || ih >= 24) continue;
#pragma unroll
      for (int kw = 0; kw < 4; ++kw) {
        int iw = ow * 2 - 1 + kw;
        if (iw < 0 || iw >= 72) continue;
        acc += xc[ih * 72 + iw] * wk[kh * 4 + kw];
      }
    }
  }
  y[idx] = acc;
}

// ---------------------------------------------------------------------------
// Per-channel batch mean / rsqrt(var+eps) over (N,H,W) -- deterministic tree
// ---------------------------------------------------------------------------
__global__ void bn_stats_kernel(const float* __restrict__ y,
                                float* __restrict__ mean, float* __restrict__ rstd) {
  const int c = blockIdx.x;            // 12 channels
  const int HW = 12 * 36;              // 432
  const int perB = 12 * HW;
  float s = 0.f, s2 = 0.f;
  for (int i = threadIdx.x; i < 512 * HW; i += blockDim.x) {
    int b = i / HW, p = i % HW;
    float v = y[(size_t)b * perB + c * HW + p];
    s += v; s2 += v * v;
  }
  __shared__ float rs[256], rq[256];
  rs[threadIdx.x] = s; rq[threadIdx.x] = s2;
  __syncthreads();
  for (int o = 128; o > 0; o >>= 1) {
    if (threadIdx.x < o) { rs[threadIdx.x] += rs[threadIdx.x + o]; rq[threadIdx.x] += rq[threadIdx.x + o]; }
    __syncthreads();
  }
  if (threadIdx.x == 0) {
    const float inv = 1.0f / (512.0f * HW);
    float m = rs[0] * inv;
    float var = rq[0] * inv - m * m;
    mean[c] = m;
    rstd[c] = rsqrtf(var + 1e-5f);
  }
}

// ---------------------------------------------------------------------------
// BN apply + ReLU + pack to f16:  sfh (512, 5184)
// ---------------------------------------------------------------------------
__global__ void bn_apply_kernel(const float* __restrict__ y,
                                const float* __restrict__ mean, const float* __restrict__ rstd,
                                const float* __restrict__ gamma, const float* __restrict__ beta,
                                _Float16* __restrict__ sfh) {
  int idx = blockIdx.x * blockDim.x + threadIdx.x;
  if (idx >= 512 * 5184) return;
  int c = (idx % 5184) / 432;
  float v = (y[idx] - mean[c]) * rstd[c] * gamma[c] + beta[c];
  sfh[idx] = (_Float16)fmaxf(v, 0.0f);
}

// src (K,N) f32 -> dst (N,K) f16    (B matrices pre-transposed for WMMA lanes)
__global__ void transpose_f32_to_f16(const float* __restrict__ src, _Float16* __restrict__ dst,
                                     int K, int N) {
  int idx = blockIdx.x * blockDim.x + threadIdx.x;
  if (idx >= K * N) return;
  int k = idx / N, n = idx % N;
  dst[(size_t)n * K + k] = (_Float16)src[idx];
}

__global__ void to_f16_kernel(const float* __restrict__ src, _Float16* __restrict__ dst, int n) {
  int i = blockIdx.x * blockDim.x + threadIdx.x;
  if (i < n) dst[i] = (_Float16)src[i];
}

// ---------------------------------------------------------------------------
// Tiny fused aux: c = adj@rowsum(adj); q_dir[n,g] = d[n]·Wih0_dir[g,:] + b;
// d[n,k] = rowsum_n*(g1b@g2W)[k] + g2b[k];  bias sums for layer-1 pre-GEMMs.
// ---------------------------------------------------------------------------
__global__ void gcn_aux_kernel(const float* __restrict__ adj,
                               const float* __restrict__ g1b, const float* __restrict__ g2W,
                               const float* __restrict__ g2b,
                               const float* __restrict__ Wih0f,
                               const float* __restrict__ bih0f, const float* __restrict__ bhh0f,
                               const float* __restrict__ Wih0b,
                               const float* __restrict__ bih0b, const float* __restrict__ bhh0b,
                               const float* __restrict__ bih1f, const float* __restrict__ bhh1f,
                               const float* __restrict__ bih1b, const float* __restrict__ bhh1b,
                               float* __restrict__ cvec,
                               float* __restrict__ q0f, float* __restrict__ q0b,
                               float* __restrict__ bsum1f, float* __restrict__ bsum1b) {
  __shared__ float srs[4];
  __shared__ float sd[1024];
  const int g = threadIdx.x;
  if (g < 4) { float s = 0.f; for (int j = 0; j < 4; ++j) s += adj[g * 4 + j]; srs[g] = s; }
  __syncthreads();
  if (g < 4) { float s = 0.f; for (int j = 0; j < 4; ++j) s += adj[g * 4 + j] * srs[j]; cvec[g] = s; }
  float gb = 0.f;                                   // GB2[g] = g1b @ g2W
  for (int k = 0; k < 1024; ++k) gb += g1b[k] * g2W[k * 256 + g];
  for (int n = 0; n < 4; ++n) sd[n * 256 + g] = srs[n] * gb + g2b[g];
  __syncthreads();
  const float b0f = bih0f[g] + bhh0f[g];
  const float b0b = bih0b[g] + bhh0b[g];
  for (int n = 0; n < 4; ++n) {
    float sf_ = b0f, sb_ = b0b;
    const float* dr = sd + n * 256;
    const float* wf = Wih0f + g * 256;
    const float* wb = Wih0b + g * 256;
    for (int k = 0; k < 256; ++k) { float dv = dr[k]; sf_ += dv * wf[k]; sb_ += dv * wb[k]; }
    q0f[n * 256 + g] = sf_;
    q0b[n * 256 + g] = sb_;
  }
  bsum1f[g] = bih1f[g] + bhh1f[g];
  bsum1b[g] = bih1b[g] + bhh1b[g];
}

// ---------------------------------------------------------------------------
// WMMA fragment loaders per CDNA5 ISA 7.12.2 (16-bit A 16x32 / B 32x16 maps)
// ---------------------------------------------------------------------------
__device__ __forceinline__ v16h load_afrag(const _Float16* __restrict__ row, int k0, int half) {
  // lane r holds row m0+r: K = {half*8 .. +7} in v0..3, {16+half*8 .. +7} in v4..7
  v8h lo = *reinterpret_cast<const v8h*>(row + k0 + half * 8);
  v8h hi = *reinterpret_cast<const v8h*>(row + k0 + 16 + half * 8);
  v16h a;
#pragma unroll
  for (int i = 0; i < 8; ++i) { a[i] = lo[i]; a[i + 8] = hi[i]; }
  return a;
}
__device__ __forceinline__ v16h load_bfrag(const _Float16* __restrict__ row, int k0, int half) {
  // lane r holds col n0+r: K = {half*8 .. +7} in v0..3, {16+half*8 .. +7} in v4..7
  v8h lo = *reinterpret_cast<const v8h*>(row + k0 + half * 8);
  v8h hi = *reinterpret_cast<const v8h*>(row + k0 + 16 + half * 8);
  v16h b;
#pragma unroll
  for (int i = 0; i < 8; ++i) { b[i] = lo[i]; b[i + 8] = hi[i]; }
  return b;
}
__device__ __forceinline__ v8f wmma_f16(v16h a, v16h b, v8f c) {
  return __builtin_amdgcn_wmma_f32_16x16x32_f16(false, a, false, b, (short)0, c, false, false);
}
__device__ __forceinline__ void store_tile(v8f acc, float* __restrict__ C, _Float16* __restrict__ Ch,
                                           const float* __restrict__ bias,
                                           int m0, int n0, int N, int r, int half) {
  const int col = n0 + r;
  const float bv = bias ? bias[col] : 0.0f;
#pragma unroll
  for (int v = 0; v < 8; ++v) {          // C/D: VGPR v -> row m0 + half*8 + v, col = lane
    const int row = m0 + (half << 3) + v;
    float val = acc[v] + bv;
    if (C)  C[(size_t)row * N + col] = val;
    if (Ch) Ch[(size_t)row * N + col] = (_Float16)val;
  }
}

// ---------------------------------------------------------------------------
// WMMA GEMM:  C(MxN,f32)[+bias] and/or Ch(f16) = A(MxK,f16) * Bt(NxK,f16)^T
// Each wave owns a 32x32 macro-tile: 2 A-frags x 2 B-frags -> 4 WMMAs per
// 4 fragment loads (2x register reuse vs 1 tile/wave). Block = 4 waves = 64x64.
// Requires M%64==0, N%64==0, K%32==0 (all call sites satisfy this).
// ---------------------------------------------------------------------------
__global__ void __launch_bounds__(128)
gemm32_wmma(const _Float16* __restrict__ A, const _Float16* __restrict__ Bt,
            float* __restrict__ C, _Float16* __restrict__ Ch,
            const float* __restrict__ bias, int M, int N, int K) {
  const int lane = threadIdx.x & 31;
  const int wave = threadIdx.x >> 5;
  const int r    = lane & 15;
  const int half = lane >> 4;
  const int m0 = (blockIdx.y << 6) + ((wave >> 1) << 5);
  const int n0 = (blockIdx.x << 6) + ((wave & 1) << 5);
  if (m0 >= M || n0 >= N) return;
  const _Float16* A0 = A + (size_t)(m0 + r) * K;
  const _Float16* A1 = A + (size_t)(m0 + 16 + r) * K;
  const _Float16* B0 = Bt + (size_t)(n0 + r) * K;
  const _Float16* B1 = Bt + (size_t)(n0 + 16 + r) * K;
  v8f acc00 = {}, acc01 = {}, acc10 = {}, acc11 = {};
  for (int k0 = 0; k0 < K; k0 += 32) {
    v16h a0 = load_afrag(A0, k0, half);
    v16h a1 = load_afrag(A1, k0, half);
    v16h b0 = load_bfrag(B0, k0, half);
    v16h b1 = load_bfrag(B1, k0, half);
    acc00 = wmma_f16(a0, b0, acc00);
    acc01 = wmma_f16(a0, b1, acc01);
    acc10 = wmma_f16(a1, b0, acc10);
    acc11 = wmma_f16(a1, b1, acc11);
  }
  store_tile(acc00, C, Ch, bias, m0,      n0,      N, r, half);
  store_tile(acc01, C, Ch, bias, m0,      n0 + 16, N, r, half);
  store_tile(acc10, C, Ch, bias, m0 + 16, n0,      N, r, half);
  store_tile(acc11, C, Ch, bias, m0 + 16, n0 + 16, N, r, half);
}

// ---------------------------------------------------------------------------
// Bidirectional LSTM layer: 2 persistent blocks (dir 0 = fwd, 1 = bwd),
// 256 threads. Whh staged transposed in LDS ([k][g], conflict-free).
// Layer-0 mode (pre==null): pre[t,n,g] = cvec[n]*P[t,g] + q[n,g]
// Layer-1 mode:             pre[t,n,g] = pre[(t*4+n)*256+g]  (bias pre-added)
// ---------------------------------------------------------------------------
__global__ void __launch_bounds__(256)
lstm_kernel(const float* __restrict__ Pf, const float* __restrict__ Pb,
            const float* __restrict__ qf, const float* __restrict__ qb,
            const float* __restrict__ pref, const float* __restrict__ preb,
            const float* __restrict__ cvec,
            const float* __restrict__ Whhf, const float* __restrict__ Whhb,
            _Float16* __restrict__ out16, float* __restrict__ out32) {
  const int dir = blockIdx.x;
  const float* P   = dir ? Pb   : Pf;
  const float* q   = dir ? qb   : qf;
  const float* pre = dir ? preb : pref;
  const float* Whh = dir ? Whhb : Whhf;
  const int g = threadIdx.x;

  __shared__ float sWhhT[64 * 256];   // [k][g] transposed: 64 KB, conflict-free
  __shared__ float sh[256], sc[256];  // h,c for 4 nodes x 64
  __shared__ float sg[1024];          // gates: 4 nodes x 256

  for (int i = g; i < 256 * 64; i += 256) {
    int gg = i >> 6, kk = i & 63;
    sWhhT[kk * 256 + gg] = Whh[i];    // Whh is (256, 64) row-major
  }
  sh[g] = 0.f; sc[g] = 0.f;
  __syncthreads();

  float cn[4];
  if (!pre) {
#pragma unroll
    for (int n = 0; n < 4; ++n) cn[n] = cvec[n];
  }

  for (int step = 0; step < 512; ++step) {
    const int t = dir ? (511 - step) : step;
    // ---- phase A: gates[n][g] = pre + h[n] . Whh[g,:] ----
    float pg[4];
    if (pre) {
#pragma unroll
      for (int n = 0; n < 4; ++n) pg[n] = pre[(size_t)(t * 4 + n) * 256 + g];
    } else {
      const float pv = P[(size_t)t * 256 + g];
#pragma unroll
      for (int n = 0; n < 4; ++n) pg[n] = cn[n] * pv + q[n * 256 + g];
    }
#pragma unroll
    for (int n = 0; n < 4; ++n) {
      const float* hr = sh + n * 64;
      float s = pg[n];
#pragma unroll 8
      for (int k = 0; k < 64; ++k) s += hr[k] * sWhhT[k * 256 + g];
      sg[n * 256 + g] = s;
    }
    __syncthreads();
    // ---- phase B: thread = (n, hh) updates c, h ----
    const int n = g >> 6, hh = g & 63;
    const float iv = sg[n * 256 + hh];
    const float fv = sg[n * 256 + 64 + hh];
    const float gv = sg[n * 256 + 128 + hh];
    const float ov = sg[n * 256 + 192 + hh];
    const float cnew = sigmf(fv) * sc[g] + sigmf(iv) * tanhf(gv);
    const float hnew = sigmf(ov) * tanhf(cnew);
    sc[g] = cnew; sh[g] = hnew;
    const size_t o = (size_t)(t * 4 + n) * 128 + dir * 64 + hh;
    if (out16) out16[o] = (_Float16)hnew;
    if (out32) out32[o] = hnew;
    __syncthreads();
  }
}

// ---------------------------------------------------------------------------
// out[b,o] = (mean_n h2[b,n,:]) @ linW + linb
// ---------------------------------------------------------------------------
__global__ void final_kernel(const float* __restrict__ h2, const float* __restrict__ linW,
                             const float* __restrict__ linb, float* __restrict__ out) {
  int idx = blockIdx.x * blockDim.x + threadIdx.x;
  if (idx >= 512 * 24) return;
  const int o = idx % 24, b = idx / 24;
  const float* hb = h2 + (size_t)b * 512;
  float s = linb[o];
  for (int f = 0; f < 128; ++f) {
    float hv = 0.25f * (hb[f] + hb[128 + f] + hb[256 + f] + hb[384 + f]);
    s += hv * linW[f * 24 + o];
  }
  out[idx] = s;
}

// ---------------------------------------------------------------------------
extern "C" void kernel_launch(void* const* d_in, const int* in_sizes, int n_in,
                              void* d_out, int out_size, void* d_ws, size_t ws_size,
                              hipStream_t stream) {
  (void)in_sizes; (void)n_in; (void)out_size; (void)ws_size;
  const float* sst   = (const float*)d_in[0];
  const float* adj   = (const float*)d_in[4];
  const float* cW    = (const float*)d_in[5];
  const float* cb    = (const float*)d_in[6];
  const float* cgam  = (const float*)d_in[7];
  const float* cbet  = (const float*)d_in[8];
  const float* g1W   = (const float*)d_in[21];
  const float* g1b   = (const float*)d_in[22];
  const float* g2W   = (const float*)d_in[23];
  const float* g2b   = (const float*)d_in[24];
  const float* Wih0f = (const float*)d_in[25]; const float* Whh0f = (const float*)d_in[26];
  const float* bih0f = (const float*)d_in[27]; const float* bhh0f = (const float*)d_in[28];
  const float* Wih0b = (const float*)d_in[29]; const float* Whh0b = (const float*)d_in[30];
  const float* bih0b = (const float*)d_in[31]; const float* bhh0b = (const float*)d_in[32];
  const float* Wih1f = (const float*)d_in[33]; const float* Whh1f = (const float*)d_in[34];
  const float* bih1f = (const float*)d_in[35]; const float* bhh1f = (const float*)d_in[36];
  const float* Wih1b = (const float*)d_in[37]; const float* Whh1b = (const float*)d_in[38];
  const float* bih1b = (const float*)d_in[39]; const float* bhh1b = (const float*)d_in[40];
  const float* linW  = (const float*)d_in[41];
  const float* linb  = (const float*)d_in[42];
  float* out = (float*)d_out;

  // ---- workspace carve-out (~36 MB) ----
  char* w = (char*)d_ws;
  auto carve = [&](size_t bytes) { char* p = w; w += (bytes + 255) & ~(size_t)255; return p; };
  float*     y      = (float*)    carve((size_t)2654208 * 4);        // conv out
  float*     mean   = (float*)    carve(64);
  float*     rstd   = (float*)    carve(64);
  _Float16*  sfh    = (_Float16*) carve((size_t)512 * 5184 * 2);     // BN+ReLU f16
  _Float16*  g1Wt   = (_Float16*) carve((size_t)1024 * 5184 * 2);
  _Float16*  g2Wt   = (_Float16*) carve((size_t)256 * 1024 * 2);
  _Float16*  S1h    = (_Float16*) carve((size_t)512 * 1024 * 2);
  _Float16*  S2h    = (_Float16*) carve((size_t)512 * 256 * 2);
  _Float16*  W0fh   = (_Float16*) carve((size_t)256 * 256 * 2);
  _Float16*  W0bh   = (_Float16*) carve((size_t)256 * 256 * 2);
  _Float16*  W1fh   = (_Float16*) carve((size_t)256 * 128 * 2);
  _Float16*  W1bh   = (_Float16*) carve((size_t)256 * 128 * 2);
  float*     cvec   = (float*)    carve(16);
  float*     q0f    = (float*)    carve(4096);
  float*     q0b    = (float*)    carve(4096);
  float*     bsum1f = (float*)    carve(1024);
  float*     bsum1b = (float*)    carve(1024);
  float*     P0f    = (float*)    carve((size_t)512 * 256 * 4);
  float*     P0b    = (float*)    carve((size_t)512 * 256 * 4);
  _Float16*  h1h    = (_Float16*) carve((size_t)2048 * 128 * 2);
  float*     pre1f  = (float*)    carve((size_t)2048 * 256 * 4);
  float*     pre1b  = (float*)    carve((size_t)2048 * 256 * 4);
  float*     h2     = (float*)    carve((size_t)2048 * 128 * 4);

  // 1) conv + BN stats + BN/ReLU + f16 pack
  conv_bias_kernel<<<2654208 / 256, 256, 0, stream>>>(sst, cW, cb, y);
  bn_stats_kernel<<<12, 256, 0, stream>>>(y, mean, rstd);
  bn_apply_kernel<<<2654208 / 256, 256, 0, stream>>>(y, mean, rstd, cgam, cbet, sfh);

  // 2) weight conversion (f16, B transposed to N x K)
  transpose_f32_to_f16<<<(5184 * 1024) / 256, 256, 0, stream>>>(g1W, g1Wt, 5184, 1024);
  transpose_f32_to_f16<<<(1024 * 256) / 256, 256, 0, stream>>>(g2W, g2Wt, 1024, 256);
  to_f16_kernel<<<256, 256, 0, stream>>>(Wih0f, W0fh, 65536);
  to_f16_kernel<<<256, 256, 0, stream>>>(Wih0b, W0bh, 65536);
  to_f16_kernel<<<128, 256, 0, stream>>>(Wih1f, W1fh, 32768);
  to_f16_kernel<<<128, 256, 0, stream>>>(Wih1b, W1bh, 32768);

  // 3) tiny aux (cvec, q, bias sums)
  gcn_aux_kernel<<<1, 256, 0, stream>>>(adj, g1b, g2W, g2b,
                                        Wih0f, bih0f, bhh0f, Wih0b, bih0b, bhh0b,
                                        bih1f, bhh1f, bih1b, bhh1b,
                                        cvec, q0f, q0b, bsum1f, bsum1b);

  // 4) GCN collapsed to two WMMA GEMMs: S1 = sf@g1W, S2 = S1@g2W
  gemm32_wmma<<<dim3(16, 8), 128, 0, stream>>>(sfh, g1Wt, nullptr, S1h, nullptr, 512, 1024, 5184);
  gemm32_wmma<<<dim3(4, 8), 128, 0, stream>>>(S1h, g2Wt, nullptr, S2h, nullptr, 512, 256, 1024);

  // 5) LSTM layer 0 input projections: P = S2 @ Wih^T (Wih is already N x K)
  gemm32_wmma<<<dim3(4, 8), 128, 0, stream>>>(S2h, W0fh, P0f, nullptr, nullptr, 512, 256, 256);
  gemm32_wmma<<<dim3(4, 8), 128, 0, stream>>>(S2h, W0bh, P0b, nullptr, nullptr, 512, 256, 256);

  // 6) LSTM layer 0 (bidirectional, persistent blocks) -> h1 (f16)
  lstm_kernel<<<2, 256, 0, stream>>>(P0f, P0b, q0f, q0b, nullptr, nullptr,
                                     cvec, Whh0f, Whh0b, h1h, nullptr);

  // 7) LSTM layer 1 input projections: pre1 = h1 @ Wih1^T + (bih+bhh)
  gemm32_wmma<<<dim3(4, 32), 128, 0, stream>>>(h1h, W1fh, pre1f, nullptr, bsum1f, 2048, 256, 128);
  gemm32_wmma<<<dim3(4, 32), 128, 0, stream>>>(h1h, W1bh, pre1b, nullptr, bsum1b, 2048, 256, 128);

  // 8) LSTM layer 1 -> h2 (f32)
  lstm_kernel<<<2, 256, 0, stream>>>(nullptr, nullptr, nullptr, nullptr, pre1f, pre1b,
                                     cvec, Whh1f, Whh1b, nullptr, h2);

  // 9) node-mean + final linear -> (512, 24)
  final_kernel<<<48, 256, 0, stream>>>(h2, linW, linb, out);
}
